// HGAT_9036611190940
// MI455X (gfx1250) — matrix operand order repaired and checked by reading, not compiled, not run on hardware
//
#include <hip/hip_runtime.h>
#include <hip/hip_bf16.h>

// ---------------------------------------------------------------------------
// Problem constants (match reference)
// ---------------------------------------------------------------------------
#define NN      50000
#define EE      800000
#define DD      256
#define HH      8
#define CC      32
#define ETD     10
#define EAD     50
#define STEPS   6
#define NEG     0.2f
#define FDIM    1024          // 4*D
#define MTILES  (NN / 16)     // 3125 exactly

typedef __attribute__((ext_vector_type(16))) __bf16 v16bf;
typedef __attribute__((ext_vector_type(8)))  float  v8f;

// ---------------------------------------------------------------------------
// Device helpers
// ---------------------------------------------------------------------------
__device__ __forceinline__ float leaky(float v) { return v > 0.f ? v : NEG * v; }

__device__ __forceinline__ void atomicMaxF(float* addr, float v) {
    if (v >= 0.f) atomicMax((int*)addr, __float_as_int(v));
    else          atomicMin((unsigned int*)addr, __float_as_uint(v));
}

// --- WMMA fragment swizzle (ISA 7.12.2 layouts) ----------------------------
// A 16x32 bf16 fragment: returns lane*16 + elem for element (row, k)
__device__ __forceinline__ int a_swz(int row, int k) {
    const int lane = (row & 15) + (((k >> 3) & 1) << 4);
    const int elem = ((((k >> 4) << 2) + ((k & 7) >> 1)) << 1) + (k & 1);
    return lane * 16 + elem;
}
// B 32x16 bf16 fragment: element (k, col)
__device__ __forceinline__ int b_swz(int k, int col) {
    const int lane = col + (((k >> 3) & 1) << 4);
    const int elem = ((k >> 4) << 3) + (k & 7);
    return lane * 16 + elem;
}
// Load one pre-swizzled fragment: 32 contiguous bytes per lane (2x b128)
__device__ __forceinline__ v16bf load_frag(const __bf16* fragbase, int fragidx, int lane) {
    return *(const v16bf*)(fragbase + ((size_t)fragidx * 32 + lane) * 16);
}

__device__ __forceinline__ v8f wmma_bf16(v16bf a, v16bf b, v8f c) {
    return __builtin_amdgcn_wmma_f32_16x16x32_bf16(false, a, false, b, (short)0, c, false, false);
}

// ---------------------------------------------------------------------------
// Utility / packing kernels
// ---------------------------------------------------------------------------
__global__ void fill_f32_kernel(float* p, float v, int n) {
    int i = blockIdx.x * 256 + threadIdx.x;
    if (i < n) p[i] = v;
}

// Pack batched row-major [B][K][N] f32 weights into B-fragment layout
// dst batch stride == K*N elements; fragment f = kk*(N/16)+nt occupies 512 elems.
__global__ void pack_B_kernel(const float* __restrict__ src, __bf16* __restrict__ dst,
                              int K, int N, int total) {
    int i = blockIdx.x * 256 + threadIdx.x;
    if (i >= total) return;
    const int kn  = K * N;
    const int b   = i / kn;
    const int rem = i - b * kn;
    const int k   = rem / N;
    const int n   = rem - k * N;
    const int frag = (k >> 5) * (N >> 4) + (n >> 4);
    dst[(size_t)b * kn + frag * 512 + b_swz(k & 31, n & 15)] = (__bf16)src[i];
}

// Pack row-major [NN][DD] f32 activations into A-fragment layout
// tile mt, kstep kk -> fragment (mt*8 + kk), 512 elems each.
__global__ void pack_A_kernel(const float* __restrict__ src, __bf16* __restrict__ dst, int n) {
    int i = blockIdx.x * 256 + threadIdx.x;
    if (i >= n) return;
    const int node = i >> 8;        // /DD
    const int d    = i & 255;
    const int frag = ((node >> 4) << 3) + (d >> 5);
    dst[(size_t)frag * 512 + a_swz(node & 15, d & 31)] = (__bf16)src[i];
}

// ---------------------------------------------------------------------------
// K1: HeteroLinear via WMMA.  x_proj[n,0:32] = m[n,:] @ Wh[type[n]] + bh[type[n]]
// One wave per 16-row tile; compute both node types, select per row.
// ---------------------------------------------------------------------------
__global__ __launch_bounds__(256)
void hetero_linear_wmma(const __bf16* __restrict__ m_sw, const __bf16* __restrict__ wh_sw,
                        const float* __restrict__ bh, const int* __restrict__ ntype,
                        float* __restrict__ xp, int numMtiles) {
    const int lane  = threadIdx.x & 31;
    const int mtile = blockIdx.x * 8 + (threadIdx.x >> 5);
    if (mtile >= numMtiles) return;

    v8f acc[2][2];
#pragma unroll
    for (int t = 0; t < 2; ++t)
#pragma unroll
        for (int j = 0; j < 2; ++j)
#pragma unroll
            for (int r = 0; r < 8; ++r) acc[t][j][r] = 0.f;

#pragma unroll
    for (int kk = 0; kk < DD / 32; ++kk) {
        const v16bf a = load_frag(m_sw, mtile * 8 + kk, lane);
#pragma unroll
        for (int t = 0; t < 2; ++t)
#pragma unroll
            for (int j = 0; j < 2; ++j) {
                // per type: frags laid out kk*2 + j, type stride = 16 frags
                const v16bf b = load_frag(wh_sw + (size_t)t * DD * CC, kk * 2 + j, lane);
                acc[t][j] = wmma_bf16(a, b, acc[t][j]);
            }
    }

    const int col0   = lane & 15;
    const int rowoff = (lane >> 4) << 3;
#pragma unroll
    for (int r = 0; r < 8; ++r) {
        const int node = mtile * 16 + r + rowoff;
        const int t    = ntype[node];
#pragma unroll
        for (int j = 0; j < 2; ++j) {
            const int c = j * 16 + col0;
            const float v = (t == 0 ? acc[0][j][r] : acc[1][j][r]) + bh[t * CC + c];
            xp[(size_t)node * CC + c] = v;
        }
    }
}

// ---------------------------------------------------------------------------
// K2: edge attention logits + segment max (pass 1 of softmax)
// ---------------------------------------------------------------------------
__global__ __launch_bounds__(256)
void edge_logits_kernel(const int* __restrict__ src, const int* __restrict__ dst,
                        const float* __restrict__ ea, const int* __restrict__ ltype,
                        const float* __restrict__ xp, const float* __restrict__ Et,
                        const float* __restrict__ Wea, const float* __restrict__ Watt,
                        float* __restrict__ logits, float* __restrict__ mx) {
    const int e = blockIdx.x * 256 + threadIdx.x;
    if (e >= EE) return;
    const int s = src[e], d = dst[e];
    const float a = ea[e];
    const int lt = ltype[e];

    float acc[HH];
#pragma unroll
    for (int h = 0; h < HH; ++h) acc[h] = 0.f;

    const float* xd = xp + (size_t)d * CC;
    const float* xs = xp + (size_t)s * CC;
#pragma unroll 4
    for (int k = 0; k < CC; ++k) {
        const float f = xd[k];
#pragma unroll
        for (int h = 0; h < HH; ++h) acc[h] += f * Watt[k * HH + h];
    }
#pragma unroll 4
    for (int k = 0; k < CC; ++k) {
        const float f = xs[k];
#pragma unroll
        for (int h = 0; h < HH; ++h) acc[h] += f * Watt[(CC + k) * HH + h];
    }
#pragma unroll
    for (int k = 0; k < ETD; ++k) {
        const float f = leaky(Et[lt * ETD + k]);
#pragma unroll
        for (int h = 0; h < HH; ++h) acc[h] += f * Watt[(2 * CC + k) * HH + h];
    }
#pragma unroll 5
    for (int k = 0; k < EAD; ++k) {
        const float f = leaky(a * Wea[k]);
#pragma unroll
        for (int h = 0; h < HH; ++h) acc[h] += f * Watt[(2 * CC + ETD + k) * HH + h];
    }
#pragma unroll
    for (int h = 0; h < HH; ++h) {
        const float l = leaky(acc[h]);
        logits[(size_t)e * HH + h] = l;
        atomicMaxF(&mx[(size_t)d * HH + h], l);
    }
}

// ---------------------------------------------------------------------------
// K3: exp(l - max) + segment sum (pass 2); exp value overwrites logits buffer
// ---------------------------------------------------------------------------
__global__ __launch_bounds__(256)
void edge_expsum_kernel(const int* __restrict__ dst, float* __restrict__ logits,
                        const float* __restrict__ mx, float* __restrict__ ssum) {
    const int e = blockIdx.x * 256 + threadIdx.x;
    if (e >= EE) return;
    const int d = dst[e];
#pragma unroll
    for (int h = 0; h < HH; ++h) {
        const float ex = __expf(logits[(size_t)e * HH + h] - mx[(size_t)d * HH + h]);
        logits[(size_t)e * HH + h] = ex;
        atomicAdd(&ssum[(size_t)d * HH + h], ex);
    }
}

// ---------------------------------------------------------------------------
// K4: alpha = exp/sum; msg = [x_src, eae] @ Wlin; scatter alpha ⊗ msg into hacc
// ---------------------------------------------------------------------------
__global__ __launch_bounds__(256)
void edge_scatter_kernel(const int* __restrict__ src, const int* __restrict__ dst,
                         const float* __restrict__ ea, const float* __restrict__ logits,
                         const float* __restrict__ ssum, const float* __restrict__ xp,
                         const float* __restrict__ Wea, const float* __restrict__ Wlin,
                         float* __restrict__ hacc) {
    const int e = blockIdx.x * 256 + threadIdx.x;
    if (e >= EE) return;
    const int s = src[e], d = dst[e];
    const float a = ea[e];

    float alpha[HH];
#pragma unroll
    for (int h = 0; h < HH; ++h)
        alpha[h] = logits[(size_t)e * HH + h] / (ssum[(size_t)d * HH + h] + 1e-16f);

    float msg[CC];
#pragma unroll
    for (int c = 0; c < CC; ++c) msg[c] = 0.f;

    const float* xs = xp + (size_t)s * CC;
#pragma unroll 2
    for (int k = 0; k < CC; ++k) {
        const float f = xs[k];
#pragma unroll
        for (int c = 0; c < CC; ++c) msg[c] += f * Wlin[k * CC + c];
    }
#pragma unroll 2
    for (int k = 0; k < EAD; ++k) {
        const float f = leaky(a * Wea[k]);
#pragma unroll
        for (int c = 0; c < CC; ++c) msg[c] += f * Wlin[(CC + k) * CC + c];
    }

    float* hd = hacc + (size_t)d * DD;
#pragma unroll
    for (int h = 0; h < HH; ++h) {
        const float al = alpha[h];
#pragma unroll
        for (int c = 0; c < CC; ++c) atomicAdd(&hd[h * CC + c], al * msg[c]);
    }
}

// ---------------------------------------------------------------------------
// K5: m' = LayerNorm(hacc + m); write fp32 row-major + A-swizzled bf16.
// One wave (32 lanes) per node; c = lane + j*32 -> kstep j, k = lane.
// ---------------------------------------------------------------------------
__global__ __launch_bounds__(256)
void ln1_kernel(const float* __restrict__ hacc, float* __restrict__ m,
                __bf16* __restrict__ m_sw, const float* __restrict__ g,
                const float* __restrict__ b) {
    const int node = blockIdx.x * 8 + (threadIdx.x >> 5);
    const int lane = threadIdx.x & 31;
    if (node >= NN) return;

    float v[8], sum = 0.f, sq = 0.f;
#pragma unroll
    for (int j = 0; j < 8; ++j) {
        const int c = lane + j * 32;
        v[j] = hacc[(size_t)node * DD + c] + m[(size_t)node * DD + c];
        sum += v[j];
        sq  += v[j] * v[j];
    }
#pragma unroll
    for (int off = 16; off > 0; off >>= 1) {
        sum += __shfl_xor(sum, off, 32);
        sq  += __shfl_xor(sq,  off, 32);
    }
    const float mean = sum * (1.f / DD);
    const float var  = sq * (1.f / DD) - mean * mean;
    const float rstd = rsqrtf(var + 1e-5f);
    const int mt  = node >> 4;
    const int row = node & 15;
    const int swz = a_swz(row, lane);   // k == lane within each kstep fragment
#pragma unroll
    for (int j = 0; j < 8; ++j) {
        const int c = lane + j * 32;
        const float o = (v[j] - mean) * rstd * g[c] + b[c];
        m[(size_t)node * DD + c] = o;
        m_sw[(size_t)(mt * 8 + j) * 512 + swz] = (__bf16)o;
    }
}

// ---------------------------------------------------------------------------
// K6: fused FFN + residual + LayerNorm2, all WMMA.
// One block (8 waves) per 16-row tile. hidden kept in LDS in A-fragment layout.
// ---------------------------------------------------------------------------
__global__ __launch_bounds__(256)
void ffn_fused_wmma(const __bf16* __restrict__ m_sw, const float* __restrict__ m_res,
                    const __bf16* __restrict__ w1, const float* __restrict__ b1,
                    const __bf16* __restrict__ w2, const float* __restrict__ b2,
                    const float* __restrict__ g2, const float* __restrict__ bb2,
                    float* __restrict__ m_out, __bf16* __restrict__ mb_out) {
    __shared__ __align__(32) __bf16 hid[(FDIM / 32) * 512];   // 32 frags, 32 KB
    __shared__ float rsum[16], rsq[16];

    const int tid  = threadIdx.x;
    const int lane = tid & 31;
    const int w    = tid >> 5;
    const int mtile = blockIdx.x;
    const int col0   = lane & 15;
    const int rowoff = (lane >> 4) << 3;

    if (tid < 16) { rsum[tid] = 0.f; rsq[tid] = 0.f; }

    // ---- GEMM1: hidden = relu(m' @ W1 + b1); wave w owns cols [w*128, w*128+128)
    v8f acc[8];
#pragma unroll
    for (int q = 0; q < 8; ++q)
#pragma unroll
        for (int r = 0; r < 8; ++r) acc[q][r] = 0.f;

#pragma unroll
    for (int kk = 0; kk < DD / 32; ++kk) {
        const v16bf a = load_frag(m_sw, mtile * 8 + kk, lane);
        // prefetch next kstep's A fragment (global_prefetch_b8)
        if (kk + 1 < DD / 32)
            __builtin_prefetch(m_sw + ((size_t)(mtile * 8 + kk + 1) * 32 + lane) * 16, 0, 1);
#pragma unroll
        for (int q = 0; q < 8; ++q) {
            const v16bf bf = load_frag(w1, kk * (FDIM / 16) + (w * 8 + q), lane);
            acc[q] = wmma_bf16(a, bf, acc[q]);
        }
    }
    // relu + bias, store into LDS in A-fragment layout (hidden is GEMM2's A).
    // element (row, col): kk2 = col>>5, within-frag k = col&31.
#pragma unroll
    for (int q = 0; q < 8; ++q) {
        const int nb = (w * 8 + q) * 16;
        const int col = nb + col0;
        const float bias = b1[col];
        const int kk2 = col >> 5;
        const int kb  = col & 31;
#pragma unroll
        for (int r = 0; r < 8; ++r) {
            const int row = r + rowoff;
            float v = acc[q][r] + bias;
            v = v > 0.f ? v : 0.f;
            hid[kk2 * 512 + a_swz(row, kb)] = (__bf16)v;
        }
    }
    __syncthreads();

    // ---- GEMM2: f = hidden @ W2 + b2; wave w owns cols [w*32, w*32+32)
    v8f acc2[2];
#pragma unroll
    for (int j = 0; j < 2; ++j)
#pragma unroll
        for (int r = 0; r < 8; ++r) acc2[j][r] = 0.f;

#pragma unroll 4
    for (int kk = 0; kk < FDIM / 32; ++kk) {
        const v16bf a = *(const v16bf*)(hid + (kk * 32 + lane) * 16);   // 2x ds_load_b128
#pragma unroll
        for (int j = 0; j < 2; ++j) {
            const v16bf bf = load_frag(w2, kk * (DD / 16) + (w * 2 + j), lane);
            acc2[j] = wmma_bf16(a, bf, acc2[j]);
        }
    }

    // ---- residual + LN2 row statistics via LDS atomics
    float vals[2][8];
#pragma unroll
    for (int j = 0; j < 2; ++j) {
        const int col = (w * 2 + j) * 16 + col0;
#pragma unroll
        for (int r = 0; r < 8; ++r) {
            const int row  = r + rowoff;
            const int node = mtile * 16 + row;
            const float v = acc2[j][r] + b2[col] + m_res[(size_t)node * DD + col];
            vals[j][r] = v;
            atomicAdd(&rsum[row], v);
            atomicAdd(&rsq[row],  v * v);
        }
    }
    __syncthreads();

#pragma unroll
    for (int j = 0; j < 2; ++j) {
        const int col = (w * 2 + j) * 16 + col0;   // col>>5 == w
        const int kb  = col & 31;
#pragma unroll
        for (int r = 0; r < 8; ++r) {
            const int row  = r + rowoff;
            const int node = mtile * 16 + row;
            const float mean = rsum[row] * (1.f / DD);
            const float var  = rsq[row] * (1.f / DD) - mean * mean;
            const float rstd = rsqrtf(var + 1e-5f);
            const float o = (vals[j][r] - mean) * rstd * g2[col] + bb2[col];
            m_out[(size_t)node * DD + col] = o;
            mb_out[(size_t)(mtile * 8 + w) * 512 + a_swz(row, kb)] = (__bf16)o;
        }
    }
}

// ---------------------------------------------------------------------------
// Host orchestration
// ---------------------------------------------------------------------------
static inline size_t alignup(size_t x) { return (x + 255) & ~(size_t)255; }

extern "C" void kernel_launch(void* const* d_in, const int* in_sizes, int n_in,
                              void* d_out, int out_size, void* d_ws, size_t ws_size,
                              hipStream_t stream) {
    (void)in_sizes; (void)n_in; (void)out_size; (void)ws_size;

    const float* x_in   = (const float*)d_in[0];
    const int*   eidx   = (const int*)d_in[1];
    const float* eattr  = (const float*)d_in[2];
    const int*   ntype  = (const int*)d_in[3];
    const int*   ltype  = (const int*)d_in[4];
    const float* Wh     = (const float*)d_in[5];
    const float* bh     = (const float*)d_in[6];
    const float* Et     = (const float*)d_in[7];
    const float* Wea    = (const float*)d_in[8];
    const float* Watt   = (const float*)d_in[9];
    const float* Wlin   = (const float*)d_in[10];
    const float* ffn_w1 = (const float*)d_in[11];
    const float* ffn_b1 = (const float*)d_in[12];
    const float* ffn_w2 = (const float*)d_in[13];
    const float* ffn_b2 = (const float*)d_in[14];
    const float* ln1_g  = (const float*)d_in[15];
    const float* ln1_b  = (const float*)d_in[16];
    const float* ln2_g  = (const float*)d_in[17];
    const float* ln2_b  = (const float*)d_in[18];

    const int* src = eidx;          // edge_index[0]
    const int* dst = eidx + EE;     // edge_index[1]

    // Workspace carve-up
    char* ws = (char*)d_ws;
    size_t off = 0;
    auto carve = [&](size_t bytes) { void* p = ws + off; off = alignup(off + bytes); return p; };
    float*  m_cur  = (float*) carve((size_t)NN * DD * 4);
    float*  hacc   = (float*) carve((size_t)NN * DD * 4);
    float*  xp     = (float*) carve((size_t)NN * CC * 4);
    float*  logits = (float*) carve((size_t)EE * HH * 4);
    float*  mx     = (float*) carve((size_t)NN * HH * 4);
    float*  ssum   = (float*) carve((size_t)NN * HH * 4);
    __bf16* m_sw   = (__bf16*)carve((size_t)NN * DD * 2);
    __bf16* wh_sw  = (__bf16*)carve((size_t)2 * DD * CC * 2);
    __bf16* w1_sw  = (__bf16*)carve((size_t)STEPS * DD * FDIM * 2);
    __bf16* w2_sw  = (__bf16*)carve((size_t)STEPS * FDIM * DD * 2);

    // One-time: m0 copy + fragment-swizzled bf16 packing of activations/weights
    hipMemcpyAsync(m_cur, x_in, (size_t)NN * DD * 4, hipMemcpyDeviceToDevice, stream);
    {
        const int nMD = NN * DD;
        pack_A_kernel<<<(nMD + 255) / 256, 256, 0, stream>>>(x_in, m_sw, nMD);
        const int nWh = 2 * DD * CC;
        pack_B_kernel<<<(nWh + 255) / 256, 256, 0, stream>>>(Wh, wh_sw, DD, CC, nWh);
        const int nW1 = STEPS * DD * FDIM;
        pack_B_kernel<<<(nW1 + 255) / 256, 256, 0, stream>>>(ffn_w1, w1_sw, DD, FDIM, nW1);
        const int nW2 = STEPS * FDIM * DD;
        pack_B_kernel<<<(nW2 + 255) / 256, 256, 0, stream>>>(ffn_w2, w2_sw, FDIM, DD, nW2);
    }

    const int edgeBlocks = (EE + 255) / 256;         // 3125
    const int nodeDBlk   = (NN * DD + 255) / 256;    // 50000
    const int nodeHBlk   = (NN * HH + 255) / 256;    // 1563

    for (int s = 0; s < STEPS; ++s) {
        // reset accumulators
        fill_f32_kernel<<<nodeDBlk, 256, 0, stream>>>(hacc, 0.f, NN * DD);
        fill_f32_kernel<<<nodeHBlk, 256, 0, stream>>>(ssum, 0.f, NN * HH);
        fill_f32_kernel<<<nodeHBlk, 256, 0, stream>>>(mx, -3.0e38f, NN * HH);

        // K1: HeteroLinear (WMMA)
        hetero_linear_wmma<<<(MTILES + 7) / 8, 256, 0, stream>>>(
            m_sw, wh_sw, bh, ntype, xp, MTILES);

        // K2..K4: edge attention (segment softmax + scatter)
        edge_logits_kernel<<<edgeBlocks, 256, 0, stream>>>(
            src, dst, eattr, ltype, xp, Et, Wea, Watt, logits, mx);
        edge_expsum_kernel<<<edgeBlocks, 256, 0, stream>>>(dst, logits, mx, ssum);
        edge_scatter_kernel<<<edgeBlocks, 256, 0, stream>>>(
            src, dst, eattr, logits, ssum, xp, Wea, Wlin, hacc);

        // K5: LN1(h + m)
        ln1_kernel<<<(NN + 7) / 8, 256, 0, stream>>>(
            hacc, m_cur, m_sw, ln1_g + s * DD, ln1_b + s * DD);

        // K6: fused FFN + residual + LN2 (WMMA); last step writes d_out
        float* out_f32 = (s == STEPS - 1) ? (float*)d_out : m_cur;
        ffn_fused_wmma<<<MTILES, 256, 0, stream>>>(
            m_sw, m_cur,
            w1_sw + (size_t)s * DD * FDIM, ffn_b1 + s * FDIM,
            w2_sw + (size_t)s * FDIM * DD, ffn_b2 + s * DD,
            ln2_g + s * DD, ln2_b + s * DD,
            out_f32, m_sw);
    }
}
